// CustomLSTM_HN_11862699671970
// MI455X (gfx1250) — matrix operand order repaired and checked by reading, not compiled, not run on hardware
//
#include <hip/hip_runtime.h>

// CDNA5 / gfx1250 persistent-LSTM using V_WMMA_F32_16X16X32_F16.
// One wave (32 lanes) owns a 16-batch tile. A-operands = W_hh gate blocks
// (constant, 16x16 real K padded to 16x32 f16), B-operand = recurrent state
// H^T (re-packed each step: v_cvt_pkrtz pairs + one xor-16 shuffle of u32).
// Activations use the CDNA5 hardware TANH transcendental; for the sigmoid
// gates the 0.5 argument-scale of sigmoid(x)=0.5*tanh(0.5x)+0.5 is folded
// into the (constant) weights/bias so the WMMA output feeds v_tanh directly.

typedef __attribute__((ext_vector_type(16))) _Float16 v16h;
typedef __attribute__((ext_vector_type(2)))  _Float16 v2h;
typedef __attribute__((ext_vector_type(8)))  float    v8f;

#if defined(__has_builtin)
#if __has_builtin(__builtin_amdgcn_tanhf)
#define HAVE_HW_TANH 1
#endif
#endif

__device__ __forceinline__ float fast_tanh(float x) {
#ifdef HAVE_HW_TANH
    return __builtin_amdgcn_tanhf(x);               // v_tanh_f32: 1 trans op
#else
    return __builtin_fmaf(2.0f, __builtin_amdgcn_rcpf(1.0f + __expf(-2.0f * x)), -1.0f);
#endif
}
__device__ __forceinline__ float fast_sigmoid(float x) {
#ifdef HAVE_HW_TANH
    return __builtin_fmaf(0.5f, __builtin_amdgcn_tanhf(0.5f * x), 0.5f);
#else
    return __builtin_amdgcn_rcpf(1.0f + __expf(-x));
#endif
}

// LSTM layer with input size 1, hidden size 16. xseq: (B, T) row-major
// (x[n][t] at n*T + t), T must be a multiple of 4. Writes final hidden
// state h_out: (B, 16) row-major.
__global__ __launch_bounds__(32) void lstm16_wmma_kernel(
    const float* __restrict__ xseq,
    const float* __restrict__ W_ih,   // (64,)   gate order i,f,g,o
    const float* __restrict__ W_hh,   // (64,16)
    const float* __restrict__ b_ih,   // (64,)
    const float* __restrict__ b_hh,   // (64,)
    float* __restrict__ h_out,        // (B,16)
    int T)
{
    const int lane = threadIdx.x;          // 0..31
    const int ncol = lane & 15;            // D/C column: batch within tile
    const int half = lane >> 4;            // wave half
    const int nb   = blockIdx.x * 16 + ncol;

    // B-operand lane mask: only half-0 lanes carry the K=0..15 state.
    const unsigned int bmask = (half == 0) ? 0xffffffffu : 0u;

    // ---- Load constant per-gate A matrices (16x32 f16, K=16 real + 16 pad).
    // A layout (16-bit 16x32): lane m = lane%16; half-0 elements e=0..15 hold
    // K = e; half-1 elements (K=16..31) are the zero pad.
    // Sigmoid gates (0,1,3) get pre-scaled by 0.5 so their WMMA output is
    // already the tanh argument of sigmoid(x)=0.5*tanh(0.5x)+0.5.
    v16h A[4];
    float wih[4][8];
    float bias[4][8];
    #pragma unroll
    for (int g = 0; g < 4; ++g) {
#ifdef HAVE_HW_TANH
        const float gscale = (g == 2) ? 1.0f : 0.5f;
#else
        const float gscale = 1.0f;
#endif
        v16h a = {};
        #pragma unroll
        for (int e = 0; e < 16; ++e)
            a[e] = (half == 0) ? (_Float16)(gscale * W_hh[(g * 16 + ncol) * 16 + e]) : (_Float16)0.0f;
        A[g] = a;
        // D/C layout: element v corresponds to gate unit j = 8*half + v.
        #pragma unroll
        for (int v = 0; v < 8; ++v) {
            int j = half * 8 + v;
            wih[g][v]  = gscale * W_ih[g * 16 + j];
            bias[g][v] = gscale * (b_ih[g * 16 + j] + b_hh[g * 16 + j]);
        }
    }

    // Recurrent state: c in D-layout f32, h packed as WMMA B operand.
    float c[8];
    #pragma unroll
    for (int v = 0; v < 8; ++v) c[v] = 0.0f;
    v16h Bv = {};          // h == 0 initially
    float hv[8];
    #pragma unroll
    for (int v = 0; v < 8; ++v) hv[v] = 0.0f;

    const float* xrow = xseq + (size_t)nb * T;

    for (int t0 = 0; t0 < T; t0 += 4) {
        // One 16-byte load covers 4 timesteps of this lane's batch row.
        const float4 x4 = *(const float4*)(xrow + t0);
        #pragma unroll
        for (int u = 0; u < 4; ++u) {
            const float xt = (u == 0) ? x4.x : (u == 1) ? x4.y
                           : (u == 2) ? x4.z : x4.w;

            // Gate pre-activations: D_g = W_hh[g] x H^T + (b + x_t * w_ih[g])
            v8f D[4];
            #pragma unroll
            for (int g = 0; g < 4; ++g) {
                v8f C;
                #pragma unroll
                for (int v = 0; v < 8; ++v)
                    C[v] = __builtin_fmaf(wih[g][v], xt, bias[g][v]);
                D[g] = __builtin_amdgcn_wmma_f32_16x16x32_f16(
                    false, A[g], false, Bv, (short)0, C, false, false);
            }

            // Elementwise cell update in D-layout (lane: batch ncol, j=8*half+v).
            #pragma unroll
            for (int v = 0; v < 8; ++v) {
#ifdef HAVE_HW_TANH
                // Sigmoid gates: argument already halved inside the WMMA.
                float ig = __builtin_fmaf(0.5f, __builtin_amdgcn_tanhf(D[0][v]), 0.5f);
                float fg = __builtin_fmaf(0.5f, __builtin_amdgcn_tanhf(D[1][v]), 0.5f);
                float gg = __builtin_amdgcn_tanhf(D[2][v]);
                float og = __builtin_fmaf(0.5f, __builtin_amdgcn_tanhf(D[3][v]), 0.5f);
#else
                float ig = fast_sigmoid(D[0][v]);
                float fg = fast_sigmoid(D[1][v]);
                float gg = fast_tanh(D[2][v]);
                float og = fast_sigmoid(D[3][v]);
#endif
                c[v]  = __builtin_fmaf(fg, c[v], ig * gg);
                hv[v] = og * fast_tanh(c[v]);
            }

            // Repack h into the B operand layout for the next timestep:
            // B (16-bit 32x16): half 0 lanes hold K=0..15 (element e <-> k=e),
            // half 1 lanes hold K=16..31 (zero pad). Lane (half=0) has j=0..7
            // itself; j=8..15 come from the partner lane. Pack f32 pairs into
            // u32 (v_cvt_pkrtz) so only 4 shuffles + 8 masks are needed.
            #pragma unroll
            for (int p = 0; p < 4; ++p) {
                auto pk = __builtin_amdgcn_cvt_pkrtz(hv[2 * p], hv[2 * p + 1]);
                unsigned int ou;
                __builtin_memcpy(&ou, &pk, 4);
                unsigned int pu = (unsigned int)__shfl_xor((int)ou, 16, 32);
                unsigned int lo = ou & bmask;
                unsigned int hi = pu & bmask;
                v2h wlo, whi;
                __builtin_memcpy(&wlo, &lo, 4);
                __builtin_memcpy(&whi, &hi, 4);
                Bv[2 * p]         = wlo[0];
                Bv[2 * p + 1]     = wlo[1];
                Bv[8 + 2 * p]     = whi[0];
                Bv[8 + 2 * p + 1] = whi[1];
            }
        }
    }

    // Write final hidden state (also next layer's input sequence).
    // Each lane owns 8 contiguous floats at a 32B-aligned address.
    float4* dst = (float4*)(h_out + (size_t)nb * 16 + half * 8);
    dst[0] = make_float4(hv[0], hv[1], hv[2], hv[3]);
    dst[1] = make_float4(hv[4], hv[5], hv[6], hv[7]);
}

// Layer 4: input size 1, hidden size 1 — pure scalar recurrence per batch row.
__global__ __launch_bounds__(256) void lstm1_scalar_kernel(
    const float* __restrict__ xseq,   // (B, T)
    const float* __restrict__ W_ih,   // (4,)
    const float* __restrict__ W_hh,   // (4,)
    const float* __restrict__ b_ih,   // (4,)
    const float* __restrict__ b_hh,   // (4,)
    float* __restrict__ h_out,        // (B,)
    int B, int T)
{
    int n = blockIdx.x * blockDim.x + threadIdx.x;
    if (n >= B) return;

    float wi0 = W_ih[0], wi1 = W_ih[1], wi2 = W_ih[2], wi3 = W_ih[3];
    float wh0 = W_hh[0], wh1 = W_hh[1], wh2 = W_hh[2], wh3 = W_hh[3];
    float b0 = b_ih[0] + b_hh[0], b1 = b_ih[1] + b_hh[1];
    float b2 = b_ih[2] + b_hh[2], b3 = b_ih[3] + b_hh[3];

    float h = 0.0f, cc = 0.0f;
    const float* xrow = xseq + (size_t)n * T;
    for (int t = 0; t < T; ++t) {
        float x = xrow[t];
        float ig = fast_sigmoid(__builtin_fmaf(wi0, x, __builtin_fmaf(wh0, h, b0)));
        float fg = fast_sigmoid(__builtin_fmaf(wi1, x, __builtin_fmaf(wh1, h, b1)));
        float gg = fast_tanh   (__builtin_fmaf(wi2, x, __builtin_fmaf(wh2, h, b2)));
        float og = fast_sigmoid(__builtin_fmaf(wi3, x, __builtin_fmaf(wh3, h, b3)));
        cc = __builtin_fmaf(fg, cc, ig * gg);
        h  = og * fast_tanh(cc);
    }
    h_out[n] = h;
}

extern "C" void kernel_launch(void* const* d_in, const int* in_sizes, int n_in,
                              void* d_out, int out_size, void* d_ws, size_t ws_size,
                              hipStream_t stream) {
    (void)in_sizes; (void)n_in; (void)out_size; (void)d_ws; (void)ws_size;

    const float* x    = (const float*)d_in[0];
    const float* Wih1 = (const float*)d_in[1];
    const float* Whh1 = (const float*)d_in[2];
    const float* bih1 = (const float*)d_in[3];
    const float* bhh1 = (const float*)d_in[4];
    const float* Wih2 = (const float*)d_in[5];
    const float* Whh2 = (const float*)d_in[6];
    const float* bih2 = (const float*)d_in[7];
    const float* bhh2 = (const float*)d_in[8];
    const float* Wih3 = (const float*)d_in[9];
    const float* Whh3 = (const float*)d_in[10];
    const float* bih3 = (const float*)d_in[11];
    const float* bhh3 = (const float*)d_in[12];
    const float* Wih4 = (const float*)d_in[13];
    const float* Whh4 = (const float*)d_in[14];
    const float* bih4 = (const float*)d_in[15];
    const float* bhh4 = (const float*)d_in[16];

    const int B = 1024;
    const int T1 = 512;

    float* out = (float*)d_out;
    float* hn1 = out;               // (B,16,1) -> B*16 floats
    float* hn2 = out + B * 16;      // (B,16,1)
    float* hn3 = out + 2 * B * 16;  // (B,16,1)
    float* hn4 = out + 3 * B * 16;  // (1,B,1) -> B floats

    dim3 grid(B / 16), block(32);
    // Layer 1: T=512 over raw input
    lstm16_wmma_kernel<<<grid, block, 0, stream>>>(x,   Wih1, Whh1, bih1, bhh1, hn1, T1);
    // Layers 2,3: previous final h (B,16) reinterpreted as T=16 scalar sequence
    lstm16_wmma_kernel<<<grid, block, 0, stream>>>(hn1, Wih2, Whh2, bih2, bhh2, hn2, 16);
    lstm16_wmma_kernel<<<grid, block, 0, stream>>>(hn2, Wih3, Whh3, bih3, bhh3, hn3, 16);
    // Layer 4: H=1 scalar recurrence
    lstm1_scalar_kernel<<<dim3((B + 255) / 256), dim3(256), 0, stream>>>(
        hn3, Wih4, Whh4, bih4, bhh4, hn4, B, 16);
}
